// Net_55224689492382
// MI455X (gfx1250) — compile-verified, matrix operand order checked by reading, not run on hardware
//
#include <hip/hip_runtime.h>

// Problem constants (from reference)
#define NN   4096
#define TT   256
#define INP  2
#define EE   64
#define HH   128
#define OUTD 2
#define GG   512   // 4*H
#define KK   192   // E + H
#define ROWS 32    // batch rows per block
#define STR  200   // padded row length (bf16 elems); 400B rows, 16B aligned

typedef __attribute__((ext_vector_type(16))) __bf16       v16bf;
typedef __attribute__((ext_vector_type(8)))  float        v8f;
typedef __attribute__((ext_vector_type(4)))  unsigned int v4u;

union FragU  { v16bf v; v4u u[2]; };
union BfPack { __bf16 b[2]; unsigned u; };

#if __has_builtin(__builtin_amdgcn_sched_group_barrier)
#define SCHED_GROUP(mask, n, id) __builtin_amdgcn_sched_group_barrier(mask, n, id)
#else
#define SCHED_GROUP(mask, n, id)
#endif
#define SG_MFMA 0x008   // MFMA/WMMA class
#define SG_DSRD 0x100   // DS read

// ---- dynamic LDS layout (bytes) ----
#define WT_OFF     0
#define WT_BYTES   (GG * STR * 2)            // 204800 : bf16 [512][200] weights (W_ih|W_hh)
#define AB_HALF    (ROWS * STR)              // elements per A panel
#define AB_OFF     (WT_OFF + WT_BYTES)
#define AB_BYTES   (2 * AB_HALF * 2)         // 25600 : double-buffered bf16 [32][200] A=[embed|h]
#define BIAS_OFF   (AB_OFF + AB_BYTES)
#define BIAS_BYTES (GG * 4)                  // 2048
#define WIN_OFF    (BIAS_OFF + BIAS_BYTES)
#define WIN_BYTES  (EE * INP * 4)            // 512
#define BIN_OFF    (WIN_OFF + WIN_BYTES)
#define BIN_BYTES  (EE * 4)                  // 256
#define WOUT_OFF   (BIN_OFF + BIN_BYTES)
#define WOUT_BYTES (OUTD * HH * 4)           // 1024
#define BOUT_OFF   (WOUT_OFF + WOUT_BYTES)
#define BOUT_BYTES (OUTD * 4)                // 8
#define SMEM_BYTES (BOUT_OFF + BOUT_BYTES)   // ~234.2 KB < 320 KB WGP LDS

__device__ __forceinline__ float fast_tanh(float x) {
#if __has_builtin(__builtin_amdgcn_tanhf)
    return __builtin_amdgcn_tanhf(x);                       // v_tanh_f32 (TRANS)
#else
    float e = __expf(-2.0f * x);
    return __builtin_amdgcn_rcpf(1.0f + e) * 2.0f - 1.0f;
#endif
}
__device__ __forceinline__ float fast_sig(float x) {
    return __builtin_amdgcn_rcpf(1.0f + __expf(-x));        // v_exp + v_rcp
}

__global__ void __launch_bounds__(256, 1)
lstm_fused_kernel(const float* __restrict__ x,
                  const float* __restrict__ W_in,  const float* __restrict__ b_in,
                  const float* __restrict__ W_ih,  const float* __restrict__ W_hh,
                  const float* __restrict__ b_ih,  const float* __restrict__ b_hh,
                  const float* __restrict__ W_out, const float* __restrict__ b_out,
                  float* __restrict__ out) {
    extern __shared__ unsigned char smem[];
    unsigned short* wt   = (unsigned short*)(smem + WT_OFF);
    unsigned short* ab   = (unsigned short*)(smem + AB_OFF);   // two panels of [32][200]
    float*          bias = (float*)(smem + BIAS_OFF);
    float*          win  = (float*)(smem + WIN_OFF);
    float*          bin  = (float*)(smem + BIN_OFF);
    float*          wout = (float*)(smem + WOUT_OFF);
    float*          bout = (float*)(smem + BOUT_OFF);

    const int tid  = threadIdx.x;
    const int lane = tid & 31;
    const int wave = tid >> 5;
    const int mt   = wave & 1;        // M tile: rows [mt*16, +16)
    const int hgrp = wave >> 1;       // hidden columns [hgrp*32, +32) within each gate
    const int lr   = lane & 15;
    const int h2   = lane >> 4;
    const int rowBase = blockIdx.x * ROWS;

    // ---- init: weights -> bf16 LDS, biases, zero h state in both panels ----
    for (int idx = tid; idx < GG * KK; idx += 256) {
        int g = idx / KK, k = idx - g * KK;
        float w = (k < EE) ? W_ih[g * EE + k] : W_hh[g * HH + (k - EE)];
        BfPack bp; bp.b[0] = (__bf16)w; bp.b[1] = (__bf16)0.f;
        wt[g * STR + k] = (unsigned short)(bp.u & 0xFFFFu);
    }
    for (int i = tid; i < GG; i += 256) bias[i] = b_ih[i] + b_hh[i];
    for (int i = tid; i < EE * INP; i += 256) win[i] = W_in[i];
    if (tid < EE)        bin[tid]  = b_in[tid];
    if (tid < OUTD * HH) wout[tid] = W_out[tid];
    if (tid < OUTD)      bout[tid] = b_out[tid];
    for (int i = tid; i < 2 * ROWS * HH; i += 256) {       // h = 0 in both panels
        int p = i >> 12, rem = i & 4095, r = rem >> 7, c = rem & 127;
        ab[p * AB_HALF + r * STR + EE + c] = 0;
    }
    __syncthreads();

    // ---- loop-invariant per-thread state (registers) ----
    // embed role: row rE = tid>>3, e-chunk e0 = (tid&7)*8
    const int rE = tid >> 3, e0 = (tid & 7) * 8;
    float winA[8], winB[8], binv[8];
    #pragma unroll
    for (int k = 0; k < 8; ++k) {
        winA[k] = win[(e0 + k) * 2];
        winB[k] = win[(e0 + k) * 2 + 1];
        binv[k] = bin[e0 + k];
    }
    const float* xrow = x + (size_t)(rowBase + rE) * TT * INP;
    unsigned short* est0 = ab + rE * STR + e0;
    unsigned short* est1 = est0 + AB_HALF;

    // GEMM role: A-row / B-row base pointers (K-half folded in)
    const unsigned short* arow0 = ab + (mt * 16 + lr) * STR + 8 * h2;
    const unsigned short* arow1 = arow0 + AB_HALF;
    const unsigned short* browf[8];   // idx = g4*2 + j
    #pragma unroll
    for (int g4 = 0; g4 < 4; ++g4)
        #pragma unroll
        for (int j = 0; j < 2; ++j)
            browf[g4 * 2 + j] = wt + (g4 * HH + hgrp * 32 + j * 16 + lr) * STR + 8 * h2;

    float biasr[8];                   // idx = g4*2 + j
    #pragma unroll
    for (int g4 = 0; g4 < 4; ++g4)
        #pragma unroll
        for (int j = 0; j < 2; ++j)
            biasr[g4 * 2 + j] = bias[g4 * HH + hgrp * 32 + j * 16 + lr];

    // h-store base: per (parity, j)
    unsigned short* hb[2][2];
    #pragma unroll
    for (int p = 0; p < 2; ++p)
        #pragma unroll
        for (int j = 0; j < 2; ++j)
            hb[p][j] = ab + p * AB_HALF + (mt * 16 + 8 * h2) * STR + EE + hgrp * 32 + j * 16 + lr;

    // output role: r = tid>>3, od = (tid>>2)&1, quarter q4 = tid&3 (32 cols each)
    const int rO = tid >> 3, od = (tid >> 2) & 1, q4 = tid & 3;
    const unsigned short* hr0 = ab + rO * STR + EE + q4 * 32;
    const unsigned short* hr1 = hr0 + AB_HALF;
    const float* wrow = wout + od * HH + q4 * 32;
    const float  boutv = bout[od];
    const float* xres = x   + (size_t)(rowBase + rO) * TT * INP + od;
    float*       ores = out + (size_t)(rowBase + rO) * TT * OUTD + od;

    float cst[2][8];
    #pragma unroll
    for (int j = 0; j < 2; ++j)
        #pragma unroll
        for (int r = 0; r < 8; ++r) cst[j][r] = 0.f;

    #pragma unroll 2
    for (int t = 0; t < TT; ++t) {
        // panel t&1 holds [embed_t | h_{t-1}] (GEMM input); panel (t+1)&1 receives h_t
        // 1) embed = relu(x_t @ W_in^T + b_in) -> current panel cols [0,64)
        {
            const float2 xv = *(const float2*)(xrow + t * INP);
            v4u pk;
            #pragma unroll
            for (int q = 0; q < 4; ++q) {
                float v0 = fmaf(xv.y, winB[2 * q],     fmaf(xv.x, winA[2 * q],     binv[2 * q]));
                float v1 = fmaf(xv.y, winB[2 * q + 1], fmaf(xv.x, winA[2 * q + 1], binv[2 * q + 1]));
                v0 = v0 > 0.f ? v0 : 0.f;
                v1 = v1 > 0.f ? v1 : 0.f;
                BfPack bp; bp.b[0] = (__bf16)v0; bp.b[1] = (__bf16)v1;
                pk[q] = bp.u;
            }
            *(v4u*)((t & 1) ? est1 : est0) = pk;
        }
        __syncthreads();   // embed_t visible; h_{t-1} visible from previous step

        // 2) gates GEMM: 48x v_wmma_f32_16x16x32_bf16, B-fragments 4-deep pipelined.
        //    Schedule pinned: [20 DS reads] then 48x [1 WMMA ; 2 DS reads] so each
        //    refill lands 4 items ahead of its consumer (waits relax to partial dscnt).
        const unsigned short* arow = (t & 1) ? arow1 : arow0;
        v16bf aF[6];
        #pragma unroll
        for (int ks = 0; ks < 6; ++ks) {
            FragU fa;
            fa.u[0] = *(const v4u*)(arow + ks * 32);
            fa.u[1] = *(const v4u*)(arow + ks * 32 + 16);
            aF[ks] = fa.v;
        }
        v8f acc[8];
        #pragma unroll
        for (int i = 0; i < 8; ++i) acc[i] = (v8f){0.f,0.f,0.f,0.f,0.f,0.f,0.f,0.f};

        FragU fb[4];
        #pragma unroll
        for (int i = 0; i < 4; ++i) {                      // prime the pipeline (items 0..3)
            const unsigned short* bp = browf[i];           // ks(i)=0 for i<8
            fb[i].u[0] = *(const v4u*)(bp);
            fb[i].u[1] = *(const v4u*)(bp + 16);
        }
        SCHED_GROUP(SG_DSRD, 20, 0);                       // aF (12) + primed B (8)
        #pragma unroll
        for (int i = 0; i < 48; ++i) {                     // item i: ks = i>>3, col idx = i&7
            acc[i & 7] = __builtin_amdgcn_wmma_f32_16x16x32_bf16(
                false, aF[i >> 3], false, fb[i & 3].v, (short)0, acc[i & 7], false, false);
            if (i + 4 < 48) {                              // refill the buffer just consumed
                const unsigned short* bp = browf[(i + 4) & 7] + ((i + 4) >> 3) * 32;
                fb[i & 3].u[0] = *(const v4u*)(bp);
                fb[i & 3].u[1] = *(const v4u*)(bp + 16);
            }
            SCHED_GROUP(SG_MFMA, 1, 0);
            SCHED_GROUP(SG_DSRD, 2, 0);
        }

        // 3) gate activations + cell update (registers); h_t -> other panel
        #pragma unroll
        for (int j = 0; j < 2; ++j) {
            unsigned short* hp = hb[(t + 1) & 1][j];
            const float bi = biasr[0 + j], bf = biasr[2 + j], bg = biasr[4 + j], bo = biasr[6 + j];
            #pragma unroll
            for (int r = 0; r < 8; ++r) {
                float iv = fast_sig (acc[0 + j][r] + bi);
                float fv = fast_sig (acc[2 + j][r] + bf);
                float gv = fast_tanh(acc[4 + j][r] + bg);
                float ov = fast_sig (acc[6 + j][r] + bo);
                float cn = fv * cst[j][r] + iv * gv;
                cst[j][r] = cn;
                float hn = ov * fast_tanh(cn);
                *(__bf16*)(hp + r * STR) = (__bf16)hn;
            }
        }
        __syncthreads();   // h_t visible

        // 4) y_t = h_t @ W_out^T + b_out + x_t : 256 threads, 32-MAC partials + shfl reduce
        {
            const unsigned short* hrow = (t & 1) ? hr0 : hr1;   // h_t panel = (t+1)&1
            float s = 0.f;
            #pragma unroll
            for (int kk = 0; kk < 4; ++kk) {
                v4u hv = *(const v4u*)(hrow + kk * 8);
                const float4 w0 = *(const float4*)(wrow + kk * 8);
                const float4 w1 = *(const float4*)(wrow + kk * 8 + 4);
                s += __uint_as_float(hv[0] << 16)         * w0.x;
                s += __uint_as_float(hv[0] & 0xFFFF0000u) * w0.y;
                s += __uint_as_float(hv[1] << 16)         * w0.z;
                s += __uint_as_float(hv[1] & 0xFFFF0000u) * w0.w;
                s += __uint_as_float(hv[2] << 16)         * w1.x;
                s += __uint_as_float(hv[2] & 0xFFFF0000u) * w1.y;
                s += __uint_as_float(hv[3] << 16)         * w1.z;
                s += __uint_as_float(hv[3] & 0xFFFF0000u) * w1.w;
            }
            s += __shfl_xor(s, 1, 32);
            s += __shfl_xor(s, 2, 32);
            if (q4 == 0) ores[t * OUTD] = s + boutv + xres[t * INP];
        }
    }
}

extern "C" void kernel_launch(void* const* d_in, const int* in_sizes, int n_in,
                              void* d_out, int out_size, void* d_ws, size_t ws_size,
                              hipStream_t stream) {
    (void)in_sizes; (void)n_in; (void)out_size; (void)d_ws; (void)ws_size;
    const float* x     = (const float*)d_in[0];
    const float* W_in  = (const float*)d_in[1];
    const float* b_in  = (const float*)d_in[2];
    const float* W_ih  = (const float*)d_in[3];
    const float* W_hh  = (const float*)d_in[4];
    const float* b_ih  = (const float*)d_in[5];
    const float* b_hh  = (const float*)d_in[6];
    const float* W_out = (const float*)d_in[7];
    const float* b_out = (const float*)d_in[8];
    float* out = (float*)d_out;

    hipFuncSetAttribute((const void*)lstm_fused_kernel,
                        hipFuncAttributeMaxDynamicSharedMemorySize, SMEM_BYTES);
    lstm_fused_kernel<<<dim3(NN / ROWS), dim3(256), SMEM_BYTES, stream>>>(
        x, W_in, b_in, W_ih, W_hh, b_ih, b_hh, W_out, b_out, out);
}
